// Dis_Net_47442208751757
// MI455X (gfx1250) — matrix operand order repaired and checked by reading, not compiled, not run on hardware
//
#include <hip/hip_runtime.h>
#include <hip/hip_bf16.h>

// ---------------------------------------------------------------------------
// Types for CDNA5 WMMA builtins
// ---------------------------------------------------------------------------
typedef __bf16 bf16_t;
typedef bf16_t v16bf __attribute__((ext_vector_type(16)));
typedef float  v8f   __attribute__((ext_vector_type(8)));

// Native f32 -> bf16 conversion (lets clang pick v_cvt_*bf16* on gfx1250
// instead of a 3-op software round-to-nearest-even sequence).
__device__ __forceinline__ bf16_t f2bf(float f) {
    return (bf16_t)f;
}
__device__ __forceinline__ unsigned short f2us(float f) {
    bf16_t h = (bf16_t)f;
    return __builtin_bit_cast(unsigned short, h);
}
__device__ __forceinline__ bf16_t us2bf(unsigned short h) {
    return __builtin_bit_cast(bf16_t, h);
}

__device__ __forceinline__ v8f wmma_bf16(v16bf a, v16bf b, v8f c) {
    return __builtin_amdgcn_wmma_f32_16x16x32_bf16(
        /*neg_a=*/false, a, /*neg_b=*/false, b,
        /*c_mod=*/(short)0, c, /*reuse_a=*/false, /*reuse_b=*/false);
}

// ---------------------------------------------------------------------------
// Conv 3x3 pad 1 + bias + relu, implicit GEMM on WMMA bf16.
// M = Cout, N = B*H*W (always a multiple of 64 in this net), K = Cin*9
// (zero padded to a multiple of 32).
// Block = 128 threads = 4 wave32s; block tile 64(co) x 64(pix); each wave
// computes a 32x32 sub-tile = 2x2 16x16 f32 accumulators -> 4 WMMAs/K-step.
// Pipeline: global->regs prefetch for tile i+1 overlaps the WMMAs of tile i;
// LDS is ping-ponged so only ONE barrier per K-step is needed; the K loop is
// unrolled by 2 with alternating register buffers (no register copies).
// ---------------------------------------------------------------------------
#define KTILE 32
#define LDP   (KTILE + 2)   // ushort padding to avoid LDS bank conflicts

__global__ __launch_bounds__(128)
void conv3x3_wmma(const float* __restrict__ in, const float* __restrict__ wgt,
                  const float* __restrict__ bias, float* __restrict__ out,
                  int Cin, int H, int W, int Cout)
{
    const int Kred = Cin * 9;
    const int HW   = H * W;
    const int pix0 = blockIdx.x * 64;
    const int co0  = blockIdx.y * 64;
    const int tid  = threadIdx.x;
    const int lane = tid & 31;
    const int wv   = tid >> 5;
    const int wm   = (wv & 1) * 32;     // wave row (cout) offset in block tile
    const int wn   = (wv >> 1) * 32;    // wave col (pixel) offset in block tile
    const int ml   = lane & 15;
    const int hm   = lane >> 4;

    __shared__ unsigned short As[2][64][LDP];   // [buf][cout][k]
    __shared__ unsigned short Bs[2][64][LDP];   // [buf][pix ][k]

    // ---- per-thread staging geometry: one half-row (16 k) of each tile ----
    const int srow  = tid >> 1;          // 0..63  (tile row: cout row / pixel)
    const int skb   = (tid & 1) * 16;    // k base within tile: 0 or 16

    // A row base (weights are row-major [Cout][Cin*9])
    const float* wrow = wgt + (size_t)(co0 + srow) * Kred;

    // B pixel coords, computed once
    const int p    = pix0 + srow;
    const int nb   = p / HW;
    const int rem  = p - nb * HW;
    const int py   = rem / W;
    const int px   = rem - py * W;
    const float* ibase = in + (size_t)nb * Cin * HW;   // + c*HW + iy*W + ix

    float av[16], bv[16], av2[16], bv2[16];

    // ---- issue global loads for one K-tile into registers ----
    auto stage = [&](int k0, float* A, float* B) {
#pragma unroll
        for (int t = 0; t < 16; ++t) {
            int k = k0 + skb + t;
            A[t] = (k < Kred) ? wrow[k] : 0.0f;
            float v = 0.0f;
            if (k < Kred) {
                int c  = k / 9;
                int rs = k - c * 9;
                int r  = rs / 3;
                int s  = rs - r * 3;
                int iy = py + r - 1, ix = px + s - 1;
                if (iy >= 0 && iy < H && ix >= 0 && ix < W)
                    v = ibase[(size_t)c * HW + iy * W + ix];
            }
            B[t] = v;
        }
    };

    v8f acc00 = {}, acc01 = {}, acc10 = {}, acc11 = {};

    // One pipeline stage: store current regs -> LDS[sel], barrier, prefetch
    // next tile into the other register set, then fragments + 4 WMMAs.
    auto body = [&](const float* cA, const float* cB, float* nA, float* nB,
                    int sel, int k0) -> bool {
#pragma unroll
        for (int t = 0; t < 16; ++t) {
            As[sel][srow][skb + t] = f2us(cA[t]);
            Bs[sel][srow][skb + t] = f2us(cB[t]);
        }
        __syncthreads();
        const bool more = (k0 + KTILE) < Kred;
        if (more) stage(k0 + KTILE, nA, nB);

        // ---- fragments (ISA 16-bit A / B layouts, wave32) ----
        v16bf a0, a1, b0, b1;
#pragma unroll
        for (int e = 0; e < 16; ++e) {
            int j  = e >> 1;
            int ka = ((j & 3) << 1) + (e & 1) + (hm << 3) + ((j >> 2) << 4);
            a0[e] = us2bf(As[sel][wm + ml][ka]);
            a1[e] = us2bf(As[sel][wm + 16 + ml][ka]);
            int kb = (e & 7) + (hm << 3) + ((e >> 3) << 4);
            b0[e] = us2bf(Bs[sel][wn + ml][kb]);
            b1[e] = us2bf(Bs[sel][wn + 16 + ml][kb]);
        }
        acc00 = wmma_bf16(a0, b0, acc00);
        acc01 = wmma_bf16(a0, b1, acc01);
        acc10 = wmma_bf16(a1, b0, acc10);
        acc11 = wmma_bf16(a1, b1, acc11);
        return more;
    };

    stage(0, av, bv);
    int k0 = 0;
    for (;;) {
        if (!body(av, bv, av2, bv2, 0, k0)) break;   // even tile -> LDS buf 0
        k0 += KTILE;
        if (!body(av2, bv2, av, bv, 1, k0)) break;   // odd tile  -> LDS buf 1
        k0 += KTILE;
    }

    // ---- epilogue: bias + relu, scatter to NCHW ----
#pragma unroll
    for (int mt = 0; mt < 2; ++mt) {
#pragma unroll
        for (int nt = 0; nt < 2; ++nt) {
            v8f acc = (mt == 0) ? (nt == 0 ? acc00 : acc01)
                                : (nt == 0 ? acc10 : acc11);
            int coB = co0 + wm + mt * 16 + hm * 8;
            int pp  = pix0 + wn + nt * 16 + ml;
            int n   = pp / HW;
            int rr  = pp - n * HW;
#pragma unroll
            for (int r = 0; r < 8; ++r) {
                int co = coB + r;
                float v = acc[r] + bias[co];
                v = fmaxf(v, 0.0f);
                out[((size_t)(n * Cout + co) * HW) + rr] = v;
            }
        }
    }
}

// ---------------------------------------------------------------------------
// FC: out[16,N] = relu?(x[16,K] @ W[N,K]^T + b[N]).  One wave per 16-col tile,
// block = 128 threads -> 64 cols per block. No LDS (B rows are contiguous in
// the row-major weight).
// ---------------------------------------------------------------------------
__global__ __launch_bounds__(128)
void fc_wmma(const float* __restrict__ x, const float* __restrict__ w,
             const float* __restrict__ bias, float* __restrict__ out,
             int N, int K, int do_relu)
{
    const int lane = threadIdx.x & 31;
    const int wv   = threadIdx.x >> 5;
    const int n0   = blockIdx.x * 64 + wv * 16;
    const int ml   = lane & 15;
    const int hm   = lane >> 4;
    const int ncol = n0 + ml;
    const bool nok = (ncol < N);

    v8f acc = {};
    for (int k0 = 0; k0 < K; k0 += 32) {
        float af[16], bf[16];
#pragma unroll
        for (int e = 0; e < 16; ++e) {
            int j  = e >> 1;
            int ka = k0 + ((j & 3) << 1) + (e & 1) + (hm << 3) + ((j >> 2) << 4);
            af[e] = (ka < K) ? x[(size_t)ml * K + ka] : 0.0f;
            int kb = k0 + (e & 7) + (hm << 3) + ((e >> 3) << 4);
            bf[e] = (nok && kb < K) ? w[(size_t)ncol * K + kb] : 0.0f;
        }
        v16bf a, b;
#pragma unroll
        for (int e = 0; e < 16; ++e) { a[e] = f2bf(af[e]); b[e] = f2bf(bf[e]); }
        acc = wmma_bf16(a, b, acc);
    }
    if (nok) {
        float bi = bias[ncol];
#pragma unroll
        for (int r = 0; r < 8; ++r) {
            int row = hm * 8 + r;                 // batch row, 0..15
            float v = acc[r] + bi;
            if (do_relu) v = fmaxf(v, 0.0f);
            out[(size_t)row * N + ncol] = v;
        }
    }
}

// ---------------------------------------------------------------------------
// 2x2 max pool, stride 2
// ---------------------------------------------------------------------------
__global__ __launch_bounds__(256)
void maxpool2(const float* __restrict__ in, float* __restrict__ out,
              int C, int H, int W, int total /* = B*C*Ho*Wo */)
{
    int i = blockIdx.x * 256 + threadIdx.x;
    if (i >= total) return;
    int Ho = H >> 1, Wo = W >> 1;
    int wo = i % Wo;
    int t  = i / Wo;
    int ho = t % Ho;
    int nc = t / Ho;                 // n*C + c
    const float* p = in + ((size_t)nc * H + 2 * ho) * W + 2 * wo;
    float v = fmaxf(fmaxf(p[0], p[1]), fmaxf(p[W], p[W + 1]));
    out[i] = v;
}

// ---------------------------------------------------------------------------
// Zero fill
// ---------------------------------------------------------------------------
__global__ __launch_bounds__(256)
void zero_fill(float* __restrict__ p, int total)
{
    int i = blockIdx.x * 256 + threadIdx.x;
    if (i < total) p[i] = 0.0f;
}

// ---------------------------------------------------------------------------
// Cluster op: per output pixel (n,y,x) with Ho=H-3, Wo=W-3:
//   diff_j = win_j - cent (cent = window index 4), dis_j = -sum_c diff^2,
//   dis_4 -= 1000, sc = softmax_j(dis), out = sum_j sc_j*diff_j + cent.
// One block per (n,y,x), 256 threads over channels.
// ---------------------------------------------------------------------------
__global__ __launch_bounds__(256)
void cluster_kernel(const float* __restrict__ in, float* __restrict__ out,
                    int C, int H, int W)
{
    const int Ho = H - 3, Wo = W - 3;
    int bid = blockIdx.x;
    int n   = bid / (Ho * Wo);
    int rem = bid - n * (Ho * Wo);
    int y   = rem / Wo;
    int x   = rem - y * Wo;
    const int tid = threadIdx.x;

    __shared__ float red[9][256];

    float part[9];
#pragma unroll
    for (int j = 0; j < 9; ++j) part[j] = 0.0f;

    for (int c = tid; c < C; c += 256) {
        const float* base = in + ((size_t)(n * C + c) * H) * W;
        float cent = base[(y + 1) * W + (x + 1)];
#pragma unroll
        for (int j = 0; j < 9; ++j) {
            int di = j / 3, dj = j - 3 * di;
            float d = base[(y + di) * W + (x + dj)] - cent;
            part[j] += d * d;
        }
    }
#pragma unroll
    for (int j = 0; j < 9; ++j) red[j][tid] = part[j];
    __syncthreads();
    for (int s = 128; s > 0; s >>= 1) {
        if (tid < s)
#pragma unroll
            for (int j = 0; j < 9; ++j) red[j][tid] += red[j][tid + s];
        __syncthreads();
    }

    float dis[9], sc[9];
#pragma unroll
    for (int j = 0; j < 9; ++j) dis[j] = -red[j][0];
    dis[4] -= 1000.0f;
    float mx = dis[0];
#pragma unroll
    for (int j = 1; j < 9; ++j) mx = fmaxf(mx, dis[j]);
    float se = 0.0f;
#pragma unroll
    for (int j = 0; j < 9; ++j) { sc[j] = __expf(dis[j] - mx); se += sc[j]; }
    float inv = 1.0f / se;
#pragma unroll
    for (int j = 0; j < 9; ++j) sc[j] *= inv;

    for (int c = tid; c < C; c += 256) {
        const float* base = in + ((size_t)(n * C + c) * H) * W;
        float cent = base[(y + 1) * W + (x + 1)];
        float o = cent;
#pragma unroll
        for (int j = 0; j < 9; ++j) {
            int di = j / 3, dj = j - 3 * di;
            o += sc[j] * (base[(y + di) * W + (x + dj)] - cent);
        }
        out[((size_t)(n * C + c) * H + y) * W + x] = o;
    }
}

// ---------------------------------------------------------------------------
// BatchNorm over (N,H,W) per channel; one block per channel.
// ---------------------------------------------------------------------------
__global__ __launch_bounds__(256)
void batchnorm_kernel(const float* __restrict__ in, float* __restrict__ out,
                      const float* __restrict__ gamma, const float* __restrict__ beta,
                      int B, int C, int HW)
{
    const int c   = blockIdx.x;
    const int tid = threadIdx.x;
    const int tot = B * HW;
    __shared__ float s1[256], s2[256];

    float a = 0.0f, b = 0.0f;
    for (int i = tid; i < tot; i += 256) {
        int n = i / HW, p = i - n * HW;
        float v = in[((size_t)(n * C + c)) * HW + p];
        a += v; b += v * v;
    }
    s1[tid] = a; s2[tid] = b;
    __syncthreads();
    for (int s = 128; s > 0; s >>= 1) {
        if (tid < s) { s1[tid] += s1[tid + s]; s2[tid] += s2[tid + s]; }
        __syncthreads();
    }
    float mu  = s1[0] / (float)tot;
    float var = s2[0] / (float)tot - mu * mu;
    float scl = rsqrtf(var + 1e-5f) * gamma[c];
    float sh  = beta[c] - mu * scl;

    for (int i = tid; i < tot; i += 256) {
        int n = i / HW, p = i - n * HW;
        size_t idx = ((size_t)(n * C + c)) * HW + p;
        out[idx] = in[idx] * scl + sh;
    }
}

// ---------------------------------------------------------------------------
// Host orchestration
// ---------------------------------------------------------------------------
extern "C" void kernel_launch(void* const* d_in, const int* in_sizes, int n_in,
                              void* d_out, int out_size, void* d_ws, size_t ws_size,
                              hipStream_t stream)
{
    const float* x = (const float*)d_in[0];
    const float* CW[13];
    const float* CB[13];
    for (int i = 0; i < 13; ++i) {
        CW[i] = (const float*)d_in[1 + 2 * i];
        CB[i] = (const float*)d_in[2 + 2 * i];
    }
    const float* gamma = (const float*)d_in[27];
    const float* beta  = (const float*)d_in[28];
    const float* fcw[3] = { (const float*)d_in[29], (const float*)d_in[31], (const float*)d_in[33] };
    const float* fcb[3] = { (const float*)d_in[30], (const float*)d_in[32], (const float*)d_in[34] };

    const int Bn = 16;
    float* bufA = (float*)d_ws;
    float* bufB = bufA + (size_t)Bn * 64 * 224 * 224;   // 205.5 MB each

    auto conv = [&](const float* in, float* out, int wi, int Cin, int H, int W, int Cout) {
        int Np = Bn * H * W;
        dim3 grid(Np / 64, Cout / 64);
        conv3x3_wmma<<<grid, dim3(128), 0, stream>>>(in, CW[wi], CB[wi], out, Cin, H, W, Cout);
    };
    auto pool = [&](const float* in, float* out, int C, int H, int W) {
        int total = Bn * C * (H / 2) * (W / 2);
        maxpool2<<<(total + 255) / 256, 256, 0, stream>>>(in, out, C, H, W, total);
    };
    auto fc = [&](const float* in, const float* w, const float* b, float* out,
                  int N, int K, int relu) {
        fc_wmma<<<(N + 63) / 64, 128, 0, stream>>>(in, w, b, out, N, K, relu);
    };

    // stage 0: 224x224
    conv(x,    bufA, 0,   3, 224, 224,  64);
    conv(bufA, bufB, 1,  64, 224, 224,  64);
    pool(bufB, bufA, 64, 224, 224);
    // stage 1: 112x112
    conv(bufA, bufB, 2,  64, 112, 112, 128);
    conv(bufB, bufA, 3, 128, 112, 112, 128);
    pool(bufA, bufB, 128, 112, 112);
    // stage 2: 56x56
    conv(bufB, bufA, 4, 128, 56, 56, 256);
    conv(bufA, bufB, 5, 256, 56, 56, 256);
    conv(bufB, bufA, 6, 256, 56, 56, 256);
    pool(bufA, bufB, 256, 56, 56);
    // stage 3: 28x28
    conv(bufB, bufA, 7, 256, 28, 28, 512);
    conv(bufA, bufB, 8, 512, 28, 28, 512);
    conv(bufB, bufA, 9, 512, 28, 28, 512);
    pool(bufA, bufB, 512, 28, 28);
    // cluster on 16x512x14x14 (bufB -> bufA, zero background)
    {
        int total = Bn * 512 * 14 * 14;
        zero_fill<<<(total + 255) / 256, 256, 0, stream>>>(bufA, total);
        int blocks = Bn * 11 * 11;
        cluster_kernel<<<blocks, 256, 0, stream>>>(bufB, bufA, 512, 14, 14);
    }
    // batchnorm (bufA -> bufB)
    batchnorm_kernel<<<512, 256, 0, stream>>>(bufA, bufB, gamma, beta, Bn, 512, 14 * 14);
    // stage 4: 14x14
    conv(bufB, bufA, 10, 512, 14, 14, 512);
    conv(bufA, bufB, 11, 512, 14, 14, 512);
    conv(bufB, bufA, 12, 512, 14, 14, 512);
    pool(bufA, bufB, 512, 14, 14);        // -> 16x512x7x7 (flat 16x25088)
    // FC head
    fc(bufB, fcw[0], fcb[0], bufA, 4096, 512 * 7 * 7, 1);
    fc(bufA, fcw[1], fcb[1], bufB, 4096, 4096, 1);
    fc(bufB, fcw[2], fcb[2], (float*)d_out, 200, 4096, 0);
}
